// Attention_59347858096503
// MI455X (gfx1250) — compile-verified, matrix operand order checked by reading, not run on hardware
//
#include <hip/hip_runtime.h>

#define D_MODEL   1024
#define NUM_HEADS 16
#define HEAD_DIM  64
#define BATCH     4
#define SEQ       1024

typedef __attribute__((ext_vector_type(16))) _Float16 v16h;
typedef __attribute__((ext_vector_type(8)))  _Float16 v8h;
typedef __attribute__((ext_vector_type(8)))  float    v8f;

// ------------------------------------------------------------------
// f32 -> f16 elementwise convert
// ------------------------------------------------------------------
__global__ void cvt_f16_kernel(const float* __restrict__ in,
                               _Float16* __restrict__ out, int n) {
  int i = blockIdx.x * blockDim.x + threadIdx.x;
  if (i < n) out[i] = (_Float16)in[i];
}

// ------------------------------------------------------------------
// W (D x D f32 row-major) -> WT f16 with WT[j][d] = W[d][j]
// 32x32 LDS tile, conflict-free, coalesced both sides
// ------------------------------------------------------------------
__global__ void transpose_cvt_kernel(const float* __restrict__ W,
                                     _Float16* __restrict__ WT) {
  __shared__ float tile[32][33];
  const int nbx = D_MODEL / 32;
  const int bx = blockIdx.x % nbx;
  const int by = blockIdx.x / nbx;
  const int tx = threadIdx.x & 31;
  const int ty = threadIdx.x >> 5;   // 0..7
#pragma unroll
  for (int k = 0; k < 32; k += 8)
    tile[ty + k][tx] = W[(size_t)(by * 32 + ty + k) * D_MODEL + bx * 32 + tx];
  __syncthreads();
#pragma unroll
  for (int k = 0; k < 32; k += 8)
    WT[(size_t)(bx * 32 + ty + k) * D_MODEL + by * 32 + tx] =
        (_Float16)tile[tx][ty + k];
}

// ------------------------------------------------------------------
// qrow[b][d] = sum_s query_in[b][s][d]
// ------------------------------------------------------------------
__global__ void qrow_kernel(const float* __restrict__ q,
                            float* __restrict__ qrow) {
  int i = blockIdx.x * blockDim.x + threadIdx.x;  // 4096
  int b = i >> 10, d = i & 1023;
  const float* p = q + (size_t)b * SEQ * D_MODEL + d;
  float s = 0.f;
  for (int t = 0; t < SEQ; ++t) s += p[(size_t)t * D_MODEL];
  qrow[i] = s;
}

// ------------------------------------------------------------------
// qsum[b][j] = qrow[b] . Wq[:,j] + SEQ * bq[j]
// ------------------------------------------------------------------
__global__ void qsum_kernel(const float* __restrict__ qrow,
                            const float* __restrict__ Wq,
                            const float* __restrict__ bq,
                            float* __restrict__ qsum) {
  int i = blockIdx.x * blockDim.x + threadIdx.x;  // 4096
  int b = i >> 10, j = i & 1023;
  const float* qr = qrow + (size_t)b * D_MODEL;
  float s = (float)SEQ * bq[j];
  for (int dd = 0; dd < D_MODEL; ++dd)
    s += qr[dd] * Wq[(size_t)dd * D_MODEL + j];
  qsum[i] = s;
}

// ------------------------------------------------------------------
// g[b][h][d] = sum_j Wk[d][h*64+j] * qsum[b][h*64+j]
// (bk contributes a per-(b,h) constant -> cancelled by softmax; dropped)
// ------------------------------------------------------------------
__global__ void gh_kernel(const float* __restrict__ Wk,
                          const float* __restrict__ qsum,
                          float* __restrict__ gh) {
  int i = blockIdx.x * blockDim.x + threadIdx.x;  // 65536
  int d = i & 1023;
  int h = (i >> 10) & 15;
  int b = i >> 14;
  const float* wrow = Wk + (size_t)d * D_MODEL + h * HEAD_DIM;
  const float* qs = qsum + (size_t)b * D_MODEL + h * HEAD_DIM;
  float s = 0.f;
#pragma unroll 8
  for (int j = 0; j < HEAD_DIM; ++j) s += wrow[j] * qs[j];
  gh[i] = s;
}

// ------------------------------------------------------------------
// WMMA GEMM helpers
// ------------------------------------------------------------------
static __device__ __forceinline__ v16h load_frag(const _Float16* p) {
  // two non-adjacent b128 loads -> one 16-half fragment
  v8h lo = *(const v8h*)(p);
  v8h hi = *(const v8h*)(p + 16);
  return __builtin_shufflevector(lo, hi, 0, 1, 2, 3, 4, 5, 6, 7,
                                 8, 9, 10, 11, 12, 13, 14, 15);
}

static __device__ __forceinline__ void store_tile(float* C, int row, int col,
                                                  int hi, int r, v8f acc,
                                                  float bval) {
  // D layout: lane l -> n = l&15, VGPR j -> m = j + 8*(l>>4)
  float* crow = C + (size_t)(row + hi * 8) * D_MODEL + (size_t)col + r;
#pragma unroll
  for (int j = 0; j < 8; ++j) crow[(size_t)j * D_MODEL] = acc[j] + bval;
}

// ------------------------------------------------------------------
// C[M,1024] = A[M,1024](f16) * B + bias, B given transposed as BT[n][k] f16.
// One wave computes a 32x32 output block = 2x2 grid of 16x16 WMMA tiles:
// 8 b128 loads feed 4 v_wmma_f32_16x16x32_f16 (2 loads/WMMA).
// A layout per ISA: lane = M index (&15), hi half selects K in
// {hi*8..hi*8+7, 16+hi*8..23+hi*8}; BT symmetric with lane = N index.
// ------------------------------------------------------------------
__global__ void wmma_gemm_bt_kernel(const _Float16* __restrict__ A,
                                    const _Float16* __restrict__ BT,
                                    const float* __restrict__ bias,
                                    float* __restrict__ C, int M) {
  const int K = D_MODEL, N = D_MODEL;
  const int lane = threadIdx.x & 31;
  const int wave = threadIdx.x >> 5;
  const int wtiles_n = N / 32;                       // 32
  int wt = blockIdx.x * (blockDim.x >> 5) + wave;
  int wm = wt / wtiles_n;
  int wn = wt - wm * wtiles_n;
  if (wm * 32 >= M) return;  // wave-uniform: EXEC stays all-ones for WMMA
  const int r = lane & 15;
  const int hi = lane >> 4;
  const _Float16* a0p = A + (size_t)(wm * 32 + r) * K + hi * 8;
  const _Float16* a1p = a0p + (size_t)16 * K;
  const _Float16* b0p = BT + (size_t)(wn * 32 + r) * K + hi * 8;
  const _Float16* b1p = b0p + (size_t)16 * K;

  v8f acc00 = {}, acc01 = {}, acc10 = {}, acc11 = {};
#pragma unroll 2
  for (int k0 = 0; k0 < K; k0 += 32) {
    v16h A0 = load_frag(a0p + k0);
    v16h A1 = load_frag(a1p + k0);
    v16h B0 = load_frag(b0p + k0);
    v16h B1 = load_frag(b1p + k0);
    acc00 = __builtin_amdgcn_wmma_f32_16x16x32_f16(false, A0, false, B0,
                                                   (short)0, acc00, false, false);
    acc01 = __builtin_amdgcn_wmma_f32_16x16x32_f16(false, A0, false, B1,
                                                   (short)0, acc01, false, false);
    acc10 = __builtin_amdgcn_wmma_f32_16x16x32_f16(false, A1, false, B0,
                                                   (short)0, acc10, false, false);
    acc11 = __builtin_amdgcn_wmma_f32_16x16x32_f16(false, A1, false, B1,
                                                   (short)0, acc11, false, false);
  }
  const float bv0 = bias[wn * 32 + r];
  const float bv1 = bias[wn * 32 + 16 + r];
  store_tile(C, wm * 32,      wn * 32,      hi, r, acc00, bv0);
  store_tile(C, wm * 32,      wn * 32 + 16, hi, r, acc01, bv1);
  store_tile(C, wm * 32 + 16, wn * 32,      hi, r, acc10, bv0);
  store_tile(C, wm * 32 + 16, wn * 32 + 16, hi, r, acc11, bv1);
}

// ------------------------------------------------------------------
// Per-(b,h) online-softmax prefix scan (flash-attention style, matches
// jax prefix-max exactly). 256 threads = 4 seq-chunks x 64 head dims.
// scores[s] = key_in[b,s,:] . g[b,h,:]; ctx written as f16.
// ------------------------------------------------------------------
__global__ void attn_scan_kernel(const float* __restrict__ key_in,
                                 const float* __restrict__ gh,
                                 const float* __restrict__ V,
                                 _Float16* __restrict__ ctx) {
  __shared__ float sc[SEQ];
  __shared__ float gsh[D_MODEL];
  __shared__ float cm[4];
  __shared__ float cden[4];
  __shared__ float cnum[4][HEAD_DIM];
  const int bh = blockIdx.x;
  const int b = bh >> 4;
  const int h = bh & 15;
  const int tid = threadIdx.x;

  for (int j = tid; j < D_MODEL; j += 256) gsh[j] = gh[(size_t)bh * D_MODEL + j];
  __syncthreads();

  // phase 0: per-key scores
  for (int s = tid; s < SEQ; s += 256) {
    const float* krow = key_in + ((size_t)b * SEQ + s) * D_MODEL;
    float acc = 0.f;
#pragma unroll 8
    for (int j = 0; j < D_MODEL; ++j) acc += krow[j] * gsh[j];
    sc[s] = acc;
  }
  __syncthreads();

  const int c = tid >> 6;   // chunk 0..3 (256 positions each)
  const int d = tid & 63;   // head dim
  const int sbeg = c * 256;
  const float* vcol =
      V + (size_t)b * SEQ * D_MODEL + (size_t)h * HEAD_DIM + d;

  // pass 1: chunk-local online softmax partials (m, den, num[d])
  float m = -__builtin_inff(), den = 0.f, num = 0.f;
  for (int s = sbeg; s < sbeg + 256; ++s) {
    float x = sc[s];
    float nm = fmaxf(m, x);
    float sc0 = __expf(m - nm);
    float p = __expf(x - nm);
    den = den * sc0 + p;
    num = num * sc0 + p * vcol[(size_t)s * D_MODEL];
    m = nm;
  }
  if (d == 0) { cm[c] = m; cden[c] = den; }
  cnum[c][d] = num;
  __syncthreads();

  // combine preceding chunks -> entering state for this chunk
  float Mr = -__builtin_inff(), Dn = 0.f, Nn = 0.f;
  for (int cc = 0; cc < c; ++cc) {
    float m2 = cm[cc], d2 = cden[cc], n2 = cnum[cc][d];
    float nm = fmaxf(Mr, m2);
    float s1 = __expf(Mr - nm), s2 = __expf(m2 - nm);
    Dn = Dn * s1 + d2 * s2;
    Nn = Nn * s1 + n2 * s2;
    Mr = nm;
  }

  // pass 2: emit per-position ctx = num/den under running prefix max
  _Float16* crow =
      ctx + (size_t)b * SEQ * D_MODEL + (size_t)h * HEAD_DIM + d;
  for (int s = sbeg; s < sbeg + 256; ++s) {
    float x = sc[s];
    float nm = fmaxf(Mr, x);
    float s1 = __expf(Mr - nm);
    float p = __expf(x - nm);
    Dn = Dn * s1 + p;
    Nn = Nn * s1 + p * vcol[(size_t)s * D_MODEL];
    Mr = nm;
    crow[(size_t)s * D_MODEL] = (_Float16)(Nn / Dn);
  }
}

// ------------------------------------------------------------------
extern "C" void kernel_launch(void* const* d_in, const int* in_sizes, int n_in,
                              void* d_out, int out_size, void* d_ws, size_t ws_size,
                              hipStream_t stream) {
  (void)in_sizes; (void)n_in; (void)out_size; (void)ws_size;
  const float* key_in   = (const float*)d_in[0];
  const float* query_in = (const float*)d_in[1];
  const float* value_in = (const float*)d_in[2];
  const float* Wk = (const float*)d_in[3];
  // bk (d_in[4]) provably cancels in the prefix softmax (constant per (b,h) row)
  const float* Wq = (const float*)d_in[5];
  const float* bq = (const float*)d_in[6];
  const float* Wv = (const float*)d_in[7];
  const float* bv = (const float*)d_in[8];
  const float* Wo = (const float*)d_in[9];
  const float* bo = (const float*)d_in[10];
  float* out = (float*)d_out;

  const size_t NTOK = (size_t)BATCH * SEQ;            // 4096 rows
  const size_t NELEM = NTOK * D_MODEL;                // 4,194,304
  char* ws = (char*)d_ws;
  size_t off = 0;
  auto carve = [&](size_t bytes) {
    char* p = ws + off;
    off = (off + bytes + 255) & ~(size_t)255;
    return p;
  };
  _Float16* value_h = (_Float16*)carve(NELEM * 2);                 // 8 MB
  _Float16* WvT     = (_Float16*)carve((size_t)D_MODEL * D_MODEL * 2);
  _Float16* WoT     = (_Float16*)carve((size_t)D_MODEL * D_MODEL * 2);
  float*    Vbuf    = (float*)   carve(NELEM * 4);                 // 16 MB
  _Float16* ctx_h   = (_Float16*)carve(NELEM * 2);                 // 8 MB
  float*    qrow    = (float*)   carve((size_t)BATCH * D_MODEL * 4);
  float*    qsum    = (float*)   carve((size_t)BATCH * D_MODEL * 4);
  float*    gh      = (float*)   carve((size_t)BATCH * NUM_HEADS * D_MODEL * 4);

  // precision/layout prep
  cvt_f16_kernel<<<(int)(NELEM / 256), 256, 0, stream>>>(value_in, value_h, (int)NELEM);
  transpose_cvt_kernel<<<(D_MODEL / 32) * (D_MODEL / 32), 256, 0, stream>>>(Wv, WvT);
  transpose_cvt_kernel<<<(D_MODEL / 32) * (D_MODEL / 32), 256, 0, stream>>>(Wo, WoT);

  // collapsed Q/K path
  qrow_kernel<<<(BATCH * D_MODEL) / 256, 256, 0, stream>>>(query_in, qrow);
  qsum_kernel<<<(BATCH * D_MODEL) / 256, 256, 0, stream>>>(qrow, Wq, bq, qsum);
  gh_kernel<<<(BATCH * NUM_HEADS * D_MODEL) / 256, 256, 0, stream>>>(Wk, qsum, gh);

  // V = value_in @ Wv + bv   (WMMA f16->f32, 32x32 per wave)
  {
    int wave_tiles = (int)(NTOK / 32) * (D_MODEL / 32);   // 4096
    wmma_gemm_bt_kernel<<<wave_tiles / 8, 256, 0, stream>>>(value_h, WvT, bv, Vbuf, (int)NTOK);
  }

  // prefix-softmax scan -> ctx (f16)
  attn_scan_kernel<<<BATCH * NUM_HEADS, 256, 0, stream>>>(key_in, gh, Vbuf, ctx_h);

  // out = ctx @ Wo + bo      (WMMA f16->f32, 32x32 per wave)
  {
    int wave_tiles = (int)(NTOK / 32) * (D_MODEL / 32);
    wmma_gemm_bt_kernel<<<wave_tiles / 8, 256, 0, stream>>>(ctx_h, WoT, bo, out, (int)NTOK);
  }
}
